// MultiHeadCausalAttention_48868137894288
// MI455X (gfx1250) — matrix operand order repaired and checked by reading, not compiled, not run on hardware
//
#include <hip/hip_runtime.h>

// ---------------------------------------------------------------------------
// MultiHeadCausalAttention for MI455X (gfx1250), wave32 + WMMA bf16.
//   out = softmax(mask(QK^T/sqrt(dh))) V @ W_proj + b_proj,  QKV = x @ W_qkv
// All matmul "B" operands are pre-transposed so every WMMA fragment load is
// a contiguous 2x b128; GEMM tiles staged via async global->LDS when available.
// ---------------------------------------------------------------------------

typedef __attribute__((ext_vector_type(16))) __bf16 v16bf;
typedef __attribute__((ext_vector_type(8)))  float  v8f;

#define BATCH 2
#define SEQ   2048
#define DIN   1024
#define DOUTT 1024
#define NH    16
#define DH    64
#define ROWS  (BATCH * SEQ)      // 4096
#define LDQKV (3 * DOUTT)        // 3072

// ---- gfx1250 async global->LDS path (guarded; fallback = plain b128) ------
#if defined(__gfx1250__) && \
    __has_builtin(__builtin_amdgcn_global_load_async_to_lds_b128) && \
    __has_builtin(__builtin_amdgcn_s_wait_asynccnt)
#define HAS_ASYNC_LDS 1
#else
#define HAS_ASYNC_LDS 0
#endif

// Builtin parameter type (from clang diagnostic): 'int __vector(4) __device__ *'
typedef int v4i __attribute__((__vector_size__(16)));
typedef __attribute__((address_space(1))) v4i glb_v4i;  // printed as "__device__"
typedef __attribute__((address_space(3))) v4i lds_v4i;  // LDS (AS3, 32-bit ptr)

__device__ __forceinline__ glb_v4i* to_glb(const void* p) {
  return (glb_v4i*)(unsigned long long)p;  // flat == AS1 numeric value
}
__device__ __forceinline__ lds_v4i* to_lds(void* p) {
  // flat LDS address: low 32 bits are the LDS byte offset (ISA 10.2 aperture)
  return (lds_v4i*)(unsigned int)(unsigned long long)p;
}

__device__ __forceinline__ void copy16B_to_lds(const void* g, void* s) {
#if HAS_ASYNC_LDS
  __builtin_amdgcn_global_load_async_to_lds_b128(to_glb(g), to_lds(s), 0, 0);
#else
  *(uint4*)s = *(const uint4*)g;
#endif
}
__device__ __forceinline__ void copy32B_to_lds(const void* g, void* s) {
#if HAS_ASYNC_LDS
  __builtin_amdgcn_global_load_async_to_lds_b128(to_glb(g), to_lds(s), 0, 0);
  __builtin_amdgcn_global_load_async_to_lds_b128(to_glb(g), to_lds(s), 16, 0);
#else
  ((uint4*)s)[0] = ((const uint4*)g)[0];
  ((uint4*)s)[1] = ((const uint4*)g)[1];
#endif
}
__device__ __forceinline__ void wait_lds_copies() {
#if HAS_ASYNC_LDS
  __builtin_amdgcn_s_wait_asynccnt(0);
#endif
}

// ---- WMMA wrapper: D = A(16x32 bf16) * B(32x16 bf16) + C(16x16 f32) -------
__device__ __forceinline__ v8f wmma_bf16(v16bf a, v16bf b, v8f c) {
  return __builtin_amdgcn_wmma_f32_16x16x32_bf16(
      /*neg_a=*/false, a, /*neg_b=*/false, b,
      /*c_mod=*/(short)0, c, /*reuse_a=*/false, /*reuse_b=*/false);
}

// ---- Fragment loaders (contiguous k order, same for A and B) ----------
// A 16x32 bf16: row = lane&15; lanes<16 hold K {0..15},
// lanes>=16 hold K {16..31}. Two contiguous 16B runs.
__device__ __forceinline__ v16bf load_a16(const __bf16* base, int lda, int lane) {
  const __bf16* p = base + (size_t)(lane & 15) * lda + ((lane >> 4) << 4);
  union { uint4 u[2]; v16bf v; } t;
  t.u[0] = *(const uint4*)(p);
  t.u[1] = *(const uint4*)(p + 8);
  return t.v;
}

// B 32x16 from a TRANSPOSED source (rows = B columns, contiguous K):
// B[k][c] = src[c][k] -> per lane one contiguous 16-element run, 2x b128.
__device__ __forceinline__ v16bf load_bT16(const __bf16* base, int ld, int k0, int lane) {
  const __bf16* p = base + (size_t)(lane & 15) * ld + k0 + ((lane >> 4) << 4);
  union { uint4 u[2]; v16bf v; } t;
  t.u[0] = *(const uint4*)(p);
  t.u[1] = *(const uint4*)(p + 8);
  return t.v;
}

// ---------------------------------------------------------------------------
// fp32 -> bf16 conversion (float4 vectorized), and convert+transpose.
// ---------------------------------------------------------------------------
__global__ void cvt_f32_to_bf16(const float* __restrict__ src,
                                __bf16* __restrict__ dst, int n4) {
  int i = blockIdx.x * blockDim.x + threadIdx.x;
  if (i < n4) {
    float4 v = ((const float4*)src)[i];
    dst[i * 4 + 0] = (__bf16)v.x;
    dst[i * 4 + 1] = (__bf16)v.y;
    dst[i * 4 + 2] = (__bf16)v.z;
    dst[i * 4 + 3] = (__bf16)v.w;
  }
}

// dst[N][K] (bf16) = src[K][N] (fp32); K == 1024 (power of two).
__global__ void cvt_t_bf16(const float* __restrict__ src,
                           __bf16* __restrict__ dst, int total, int C) {
  int i = blockIdx.x * blockDim.x + threadIdx.x;  // i = n*1024 + k
  if (i < total) {
    int k = i & 1023;
    int n = i >> 10;
    dst[i] = (__bf16)src[(size_t)k * C + n];
  }
}

// vt[(b*NH+h)*DH + f][n] = qkv[(b*SEQ+n)*LDQKV + 2*DOUTT + h*DH + f]
__global__ void transpose_v_bf16(const __bf16* __restrict__ qkv,
                                 __bf16* __restrict__ vt) {
  int i = blockIdx.x * blockDim.x + threadIdx.x;  // 4M: n | f | bh
  int n = i & (SEQ - 1);
  int f = (i >> 11) & (DH - 1);
  int bh = i >> 17;
  int b = bh >> 4, h = bh & 15;
  vt[i] = qkv[(size_t)(b * SEQ + n) * LDQKV + 2 * DOUTT + h * DH + f];
}

// ---------------------------------------------------------------------------
// bf16 GEMM: C[M,N] = A[M,K] @ BT[N,K]^T ; block 128x64, 8 waves of 32x32,
// K-step 32. LDS-staged (async when available); fp32 accumulate in WMMA.
// ---------------------------------------------------------------------------
template <bool BF16_OUT, bool ADD_BIAS>
__global__ __launch_bounds__(256) void gemm_bf16_kernel(
    const __bf16* __restrict__ A, const __bf16* __restrict__ BT,
    __bf16* __restrict__ Cb, float* __restrict__ Cf,
    const float* __restrict__ bias, int M, int N, int K) {
  constexpr int BM = 128, BN = 64, BK = 32;
  constexpr int LDS_LD = BK + 8;  // 40 elems (80B rows, 16B aligned)
  __shared__ __attribute__((aligned(16))) __bf16 sA[BM * LDS_LD];
  __shared__ __attribute__((aligned(16))) __bf16 sBT[BN * LDS_LD];

  const int tid  = threadIdx.x;
  const int lane = tid & 31;
  const int wave = tid >> 5;
  const int wr = (wave >> 1) << 5;  // 0,32,64,96
  const int wc = (wave & 1) << 5;   // 0,32
  const int bm = blockIdx.x * BM;
  const int bn = blockIdx.y * BN;

  v8f acc[2][2];
#pragma unroll
  for (int i = 0; i < 2; ++i)
#pragma unroll
    for (int j = 0; j < 2; ++j)
#pragma unroll
      for (int e = 0; e < 8; ++e) acc[i][j][e] = 0.0f;

  const int ar = tid >> 1, ac = (tid & 1) << 4;  // A: 128x32, 16 elem/thread
  const int br = tid >> 2, bc = (tid & 3) << 3;  // BT: 64x32,  8 elem/thread

  for (int k0 = 0; k0 < K; k0 += BK) {
    copy32B_to_lds(A  + (size_t)(bm + ar) * K + k0 + ac, sA  + ar * LDS_LD + ac);
    copy16B_to_lds(BT + (size_t)(bn + br) * K + k0 + bc, sBT + br * LDS_LD + bc);
    if (k0 + BK < K) {  // global_prefetch_b8 next K-slab
      __builtin_prefetch(A  + (size_t)(bm + ar) * K + k0 + BK + ac, 0, 0);
      __builtin_prefetch(BT + (size_t)(bn + br) * K + k0 + BK + bc, 0, 0);
    }
    wait_lds_copies();
    __syncthreads();

    v16bf af0 = load_a16(sA + (size_t)wr * LDS_LD, LDS_LD, lane);
    v16bf af1 = load_a16(sA + (size_t)(wr + 16) * LDS_LD, LDS_LD, lane);
    v16bf bf0 = load_bT16(sBT + (size_t)wc * LDS_LD, LDS_LD, 0, lane);
    v16bf bf1 = load_bT16(sBT + (size_t)(wc + 16) * LDS_LD, LDS_LD, 0, lane);
    acc[0][0] = wmma_bf16(af0, bf0, acc[0][0]);
    acc[0][1] = wmma_bf16(af0, bf1, acc[0][1]);
    acc[1][0] = wmma_bf16(af1, bf0, acc[1][0]);
    acc[1][1] = wmma_bf16(af1, bf1, acc[1][1]);
    __syncthreads();
  }

  const int rbase = (lane >> 4) << 3;  // C/D: M = vgpr + 8*(lane>=16)
  const int cc = lane & 15;
#pragma unroll
  for (int i = 0; i < 2; ++i)
#pragma unroll
    for (int j = 0; j < 2; ++j) {
      const int row = bm + wr + (i << 4) + rbase;
      const int col = bn + wc + (j << 4) + cc;
#pragma unroll
      for (int r = 0; r < 8; ++r) {
        float v = acc[i][j][r];
        if (ADD_BIAS) v += bias[col];
        if (BF16_OUT) Cb[(size_t)(row + r) * N + col] = (__bf16)v;
        else          Cf[(size_t)(row + r) * N + col] = v;
      }
    }
}

// ---------------------------------------------------------------------------
// Flash-attention: one wave per 16-row Q tile, 32 keys per iteration.
// Online softmax fp32; P transposed C->A layout via per-wave LDS bounce.
// K^T frags from row-major K; V frags from pre-transposed V: all 2x b128.
// ---------------------------------------------------------------------------
__global__ __launch_bounds__(128) void attn_kernel(
    const __bf16* __restrict__ qkv, const __bf16* __restrict__ vt,
    __bf16* __restrict__ ctx) {
  constexpr int LDP = 40;  // 16x32 P tile, padded rows (80B, 16B aligned)
  __shared__ __attribute__((aligned(16))) __bf16 pbuf[4][16 * LDP];

  const int lane = threadIdx.x & 31;
  const int wave = threadIdx.x >> 5;
  const int bh = blockIdx.x;       // b*NH + h
  const int b = bh >> 4;
  const int h = bh & 15;
  const int qt = blockIdx.y * 4 + wave;
  const int row0 = qt << 4;

  const __bf16* qp  = qkv + (size_t)b * SEQ * LDQKV + h * DH;
  const __bf16* kp  = qp + DOUTT;
  const __bf16* vth = vt + (size_t)bh * DH * SEQ;  // [DH][SEQ]

  // Q tile 16 x 64 -> two A fragments (K blocks 0..31, 32..63)
  v16bf qf0 = load_a16(qp + (size_t)row0 * LDQKV + 0,  LDQKV, lane);
  v16bf qf1 = load_a16(qp + (size_t)row0 * LDQKV + 32, LDQKV, lane);

  float m[8], l[8];
  v8f acc[4];  // ctx 16 x 64 fp32
#pragma unroll
  for (int r = 0; r < 8; ++r) { m[r] = -1e30f; l[r] = 0.0f; }
#pragma unroll
  for (int j = 0; j < 4; ++j)
#pragma unroll
    for (int e = 0; e < 8; ++e) acc[j][e] = 0.0f;

  const float scale = 0.125f;  // 1/sqrt(64)
  const int col = lane & 15;
  const int rbase = (lane >> 4) << 3;
  __bf16* pb = &pbuf[wave][0];
  const int kend = row0 + 16;  // causal

  for (int kb = 0; kb < kend; kb += 32) {
    // S = Q K^T for 32 keys -> two 16x16 fp32 fragments
    v8f s0, s1;
#pragma unroll
    for (int e = 0; e < 8; ++e) { s0[e] = 0.0f; s1[e] = 0.0f; }
    v16bf b00 = load_bT16(kp + (size_t)kb * LDQKV,        LDQKV, 0,  lane);
    v16bf b01 = load_bT16(kp + (size_t)(kb + 16) * LDQKV, LDQKV, 0,  lane);
    s0 = wmma_bf16(qf0, b00, s0);
    s1 = wmma_bf16(qf0, b01, s1);
    v16bf b10 = load_bT16(kp + (size_t)kb * LDQKV,        LDQKV, 32, lane);
    v16bf b11 = load_bT16(kp + (size_t)(kb + 16) * LDQKV, LDQKV, 32, lane);
    s0 = wmma_bf16(qf1, b10, s0);
    s1 = wmma_bf16(qf1, b11, s1);

    // Online softmax per row (rows 0-7 in lanes 0-15, rows 8-15 in 16-31).
#pragma unroll
    for (int r = 0; r < 8; ++r) {
      const int qi = row0 + rbase + r;
      float v0 = (kb + col <= qi)      ? s0[r] * scale : -1e30f;
      float v1 = (kb + 16 + col <= qi) ? s1[r] * scale : -1e30f;
      float mx = fmaxf(v0, v1);
      mx = fmaxf(mx, __shfl_xor(mx, 8, 32));
      mx = fmaxf(mx, __shfl_xor(mx, 4, 32));
      mx = fmaxf(mx, __shfl_xor(mx, 2, 32));
      mx = fmaxf(mx, __shfl_xor(mx, 1, 32));
      const float mn = fmaxf(m[r], mx);
      const float p0 = __expf(v0 - mn);
      const float p1 = __expf(v1 - mn);
      float rs = p0 + p1;
      rs += __shfl_xor(rs, 8, 32);
      rs += __shfl_xor(rs, 4, 32);
      rs += __shfl_xor(rs, 2, 32);
      rs += __shfl_xor(rs, 1, 32);
      const float corr = __expf(m[r] - mn);
      l[r] = l[r] * corr + rs;
      m[r] = mn;
#pragma unroll
      for (int j = 0; j < 4; ++j) acc[j][r] *= corr;
      pb[(rbase + r) * LDP + col]      = (__bf16)p0;  // LDS in-order per wave
      pb[(rbase + r) * LDP + col + 16] = (__bf16)p1;
    }

    // ctx += P(16x32) @ V(32x64); V frags contiguous from transposed V
    v16bf pf = load_a16(pb, LDP, lane);
#pragma unroll
    for (int j = 0; j < 4; ++j) {
      v16bf vf = load_bT16(vth + (size_t)(j << 4) * SEQ + kb, SEQ, 0, lane);
      acc[j] = wmma_bf16(pf, vf, acc[j]);
    }
  }

  // Normalize and store ctx tile (bf16, [ROWS, D_OUT], head slice)
  __bf16* cp = ctx + (size_t)(b * SEQ + row0) * DOUTT + h * DH;
#pragma unroll
  for (int j = 0; j < 4; ++j)
#pragma unroll
    for (int r = 0; r < 8; ++r)
      cp[(size_t)(rbase + r) * DOUTT + (j << 4) + col] = (__bf16)(acc[j][r] / l[r]);
}

// ---------------------------------------------------------------------------
extern "C" void kernel_launch(void* const* d_in, const int* in_sizes, int n_in,
                              void* d_out, int out_size, void* d_ws, size_t ws_size,
                              hipStream_t stream) {
  const float* x     = (const float*)d_in[0];
  const float* Wqkv  = (const float*)d_in[1];
  const float* Wproj = (const float*)d_in[2];
  const float* bproj = (const float*)d_in[3];
  float* out = (float*)d_out;

  // Workspace layout (bf16), total ~56 MB
  __bf16* xb    = (__bf16*)d_ws;                      // 4096 x 1024
  __bf16* wqbT  = xb    + (size_t)ROWS * DIN;         // 3072 x 1024 (W_qkv^T)
  __bf16* wpbT  = wqbT  + (size_t)3 * DOUTT * DIN;    // 1024 x 1024 (W_proj^T)
  __bf16* qkvb  = wpbT  + (size_t)DOUTT * DOUTT;      // 4096 x 3072
  __bf16* ctxb  = qkvb  + (size_t)ROWS * 3 * DOUTT;   // 4096 x 1024
  __bf16* vtb   = ctxb  + (size_t)ROWS * DOUTT;       // 32 x 64 x 2048 (V^T)

  const int T = 256;
  cvt_f32_to_bf16<<<(ROWS * DIN / 4) / T, T, 0, stream>>>(x, xb, ROWS * DIN / 4);
  cvt_t_bf16<<<(3 * DOUTT * DIN) / T, T, 0, stream>>>(Wqkv, wqbT, 3 * DOUTT * DIN, 3 * DOUTT);
  cvt_t_bf16<<<(DOUTT * DIN) / T, T, 0, stream>>>(Wproj, wpbT, DOUTT * DIN, DOUTT);

  // QKV = x @ W_qkv   (bf16 out)
  gemm_bf16_kernel<true, false><<<dim3(ROWS / 128, 3 * DOUTT / 64), 256, 0, stream>>>(
      xb, wqbT, qkvb, nullptr, nullptr, ROWS, 3 * DOUTT, DIN);

  // V^T for contiguous P@V fragments
  transpose_v_bf16<<<(ROWS * DOUTT) / T, T, 0, stream>>>(qkvb, vtb);

  // Flash attention per (batch, head, 4 q-tiles per block)
  attn_kernel<<<dim3(BATCH * NH, SEQ / 64), 128, 0, stream>>>(qkvb, vtb, ctxb);

  // out = ctx @ W_proj + b_proj   (fp32 out, fused bias)
  gemm_bf16_kernel<false, true><<<dim3(ROWS / 128, DOUTT / 64), 256, 0, stream>>>(
      ctxb, wpbT, nullptr, out, bproj, ROWS, DOUTT, DIN);
}
